// Attention_18502719111471
// MI455X (gfx1250) — matrix operand order repaired and checked by reading, not compiled, not run on hardware
//
#include <hip/hip_runtime.h>
#include <math.h>

// ---------------------------------------------------------------------------
// Problem constants (reference: b=1, l=512, D_ORIG=64, N_HEAD=8, D_HEAD=32)
// ---------------------------------------------------------------------------
#define SEQ     512
#define D_ORIG  64
#define D_EMB   256
#define N_HEAD  8
#define D_HEAD  32
#define Q_SCALE 0.17677669529663687f   // 1/sqrt(D_HEAD)

typedef __attribute__((ext_vector_type(16))) __bf16        v16bf;
typedef __attribute__((ext_vector_type(8)))  float         v8f;
typedef __attribute__((ext_vector_type(4)))  unsigned int  u32x4;
typedef __attribute__((ext_vector_type(8)))  int           i32x8;
typedef __attribute__((ext_vector_type(4)))  int           i32x4;

// ---------------------------------------------------------------------------
// WMMA fragment builders (CDNA5 16-bit layouts, CDNA5 ISA §7.12.2)
//
// A 16x32 bf16: lane L: row M = L&15, g = L>>4;
//   e=0..7  -> K = kbase + g*8 + e ; e=8..15 -> K = kbase + 16 + g*8 + (e-8)
// B 32x16 bf16: lane L holds col N = L&15; elements follow A's K order
// C/D 16x16 f32: element r -> row M = r + ((L>>4)<<3), col N = L&15
// ---------------------------------------------------------------------------
__device__ inline v16bf load_a_frag_f32(const float* __restrict__ base, int ld,
                                        int row0, int kbase, int lane) {
  const int r = row0 + (lane & 15);
  const int g = (lane >> 4) & 1;
  const float* p = base + (size_t)r * ld + kbase + g * 8;
  v16bf a;
#pragma unroll
  for (int e = 0; e < 8; ++e)  a[e]     = (__bf16)p[e];
#pragma unroll
  for (int e = 0; e < 8; ++e)  a[8 + e] = (__bf16)p[16 + e];
  return a;
}

// Same layout, but source rows live in LDS (e_lds is 512x64 row-major).
__device__ inline v16bf load_a_frag_lds(const float* base, int row0, int kbase,
                                        int lane) {
  const int r = row0 + (lane & 15);
  const int g = (lane >> 4) & 1;
  const float* p = base + r * D_ORIG + kbase + g * 8;
  v16bf a;
#pragma unroll
  for (int e = 0; e < 8; ++e)  a[e]     = (__bf16)p[e];
#pragma unroll
  for (int e = 0; e < 8; ++e)  a[8 + e] = (__bf16)p[16 + e];
  return a;
}

__device__ inline v16bf load_b_frag_f32(const float* __restrict__ base, int ld,
                                        int kbase, int col0, int lane) {
  const int g = (lane >> 4) & 1;
  const float* p = base + (size_t)kbase * ld + col0 + (lane & 15);
  v16bf b;
#pragma unroll
  for (int e = 0; e < 8; ++e)  b[e]     = (__bf16)p[(size_t)(g * 8 + e) * ld];
#pragma unroll
  for (int e = 0; e < 8; ++e)  b[8 + e] = (__bf16)p[(size_t)(16 + g * 8 + e) * ld];
  return b;
}

__device__ inline v8f wmma_bf16(v16bf a, v16bf b, v8f c) {
  return __builtin_amdgcn_wmma_f32_16x16x32_bf16(false, a, false, b,
                                                 (short)0, c, false, false);
}

// ---------------------------------------------------------------------------
// TDM helper: build a 2-D tensor descriptor (D#) and issue TENSOR_LOAD_TO_LDS.
// Layout per CDNA5 ISA:
//   group0: [1:0]=count=1, [63:32]=lds_addr, [120:64]=global_addr,
//           [127:126]=type=2
//   group1: [17:16]=data_size(2 -> 4B), [79:48]=tensor_dim0,
//           [111:80]=tensor_dim1, [127:112]=tile_dim0, [143:128]=tile_dim1,
//           [207:160]=tensor_dim0_stride
// This toolchain exposes the 6-arg builtin:
//   (u32x4 g0, i32x8 g1, i32x4, i32x4, i32x8, i32 cpol)
// ---------------------------------------------------------------------------
__device__ inline unsigned lds_offset_of(const void* p) {
  return (unsigned)(uintptr_t)(__attribute__((address_space(3))) const void*)p;
}

__device__ inline void tdm_load_2d(unsigned lds_addr, const void* gptr,
                                   unsigned rows, unsigned cols,
                                   unsigned row_stride_elems) {
  const unsigned long long ga = (unsigned long long)(uintptr_t)gptr;
  u32x4 g0;
  g0.x = 1u;                                            // count = 1
  g0.y = lds_addr;
  g0.z = (unsigned)(ga & 0xFFFFFFFFu);
  g0.w = ((unsigned)(ga >> 32) & 0x01FFFFFFu) | 0x80000000u;   // type = 2
  i32x8 g1;
  g1[0] = (int)(2u << 16);                              // data_size = 4 bytes
  g1[1] = (int)((cols & 0xFFFFu) << 16);                // tensor_dim0[15:0]
  g1[2] = (int)(((cols >> 16) & 0xFFFFu) |              // tensor_dim0[31:16]
                ((rows & 0xFFFFu) << 16));              // tensor_dim1[15:0]
  g1[3] = (int)(((rows >> 16) & 0xFFFFu) |              // tensor_dim1[31:16]
                ((cols & 0xFFFFu) << 16));              // tile_dim0
  g1[4] = (int)(rows & 0xFFFFu);                        // tile_dim1
  g1[5] = (int)row_stride_elems;                        // dim0_stride[31:0]
  g1[6] = 0;                                            // stride hi / d1stride
  g1[7] = 0;
  const i32x4 gz4 = {0, 0, 0, 0};
  const i32x8 gz8 = {0, 0, 0, 0, 0, 0, 0, 0};
  __builtin_amdgcn_tensor_load_to_lds(g0, g1, gz4, gz4, gz8, 0);
}

// ---------------------------------------------------------------------------
// Kernel 1: Q/K/V projections.  x(512x64) @ W(64x256).
// Q stored row-major [512][256] (scaled by 1/sqrt(D_HEAD));
// K,V stored TRANSPOSED [256][512] so the attention kernel's per-column
// reads of 8 consecutive j become two global_load_b128.
// ---------------------------------------------------------------------------
__global__ __launch_bounds__(32)
void qkv_proj_kernel(const float* __restrict__ x,
                     const float* __restrict__ Wq,
                     const float* __restrict__ Wk,
                     const float* __restrict__ Wv,
                     float* __restrict__ Q, float* __restrict__ KT,
                     float* __restrict__ VT) {
  const int lane  = threadIdx.x;
  const int mt    = blockIdx.x;
  const int nt    = blockIdx.y;
  const int which = blockIdx.z;
  const float* W  = (which == 0) ? Wq : (which == 1) ? Wk : Wv;
  float* Out      = (which == 0) ? Q  : (which == 1) ? KT : VT;

  v8f acc = {};
  v16bf a0 = load_a_frag_f32(x, D_ORIG, mt * 16, 0,  lane);
  v16bf b0 = load_b_frag_f32(W, D_EMB,  0,  nt * 16, lane);
  acc = wmma_bf16(a0, b0, acc);
  v16bf a1 = load_a_frag_f32(x, D_ORIG, mt * 16, 32, lane);
  v16bf b1 = load_b_frag_f32(W, D_EMB,  32, nt * 16, lane);
  acc = wmma_bf16(a1, b1, acc);

  const int col   = nt * 16 + (lane & 15);
  const int rbase = mt * 16 + ((lane >> 4) << 3);
  if (which == 0) {
#pragma unroll
    for (int r = 0; r < 8; ++r)
      Out[(size_t)(rbase + r) * D_EMB + col] = acc[r] * Q_SCALE;
  } else {
#pragma unroll
    for (int r = 0; r < 8; ++r)
      Out[(size_t)col * SEQ + rbase + r] = acc[r];
  }
}

// ---------------------------------------------------------------------------
// Kernel 2: fused edge attention, one query row i per 128-thread block.
// e[i] (512x64 fp32, 128 KB) is DMA'd into LDS by the Tensor Data Mover in
// two 64 KB halves (double buffered on TENSORcnt); all 4 waves build WMMA
// A-fragments from LDS.  Wave w owns heads {2w, 2w+1}; the loop-invariant
// We B-fragments and q scalars are hoisted out of the j-loop.  Online
// softmax; ee tiles never leave the fp32 C fragments.
// ---------------------------------------------------------------------------
__global__ __launch_bounds__(128)
void edge_attn_kernel(const float* __restrict__ e,
                      const float* __restrict__ We,
                      const float* __restrict__ Q,
                      const float* __restrict__ KT,
                      const float* __restrict__ VT,
                      float* __restrict__ Oatt) {
  extern __shared__ float smem[];
  float* e_lds = smem;                       // SEQ x D_ORIG = 32768 floats
  float* qs    = smem + SEQ * D_ORIG;        // D_EMB floats

  const int i    = blockIdx.x;
  const int tid  = threadIdx.x;
  const int lane = tid & 31;
  const int wave = tid >> 5;                 // 0..3
  const int h0   = wave * 2;                 // first head owned by this wave
  const int rbase = (lane >> 4) << 3;        // rows 0-7 or 8-15 of a tile
  const int cl    = lane & 15;

  for (int t = tid; t < D_EMB; t += 128) qs[t] = Q[(size_t)i * D_EMB + t];

  // ---- TDM: stage e[i] into LDS, two halves, double buffered -------------
  const float* egbl = e + (size_t)i * SEQ * D_ORIG;
  if (wave == 0) {
    const unsigned lds0 = lds_offset_of(e_lds);
    const unsigned half = (SEQ / 2) * D_ORIG * sizeof(float);   // 64 KB
    tdm_load_2d(lds0,        egbl,                      SEQ / 2, D_ORIG, D_ORIG);
    tdm_load_2d(lds0 + half, egbl + (SEQ / 2) * D_ORIG, SEQ / 2, D_ORIG, D_ORIG);
    __builtin_amdgcn_s_wait_tensorcnt(1);    // first half landed
  }
  __syncthreads();

  // ---- loop-invariant B fragments (We columns of this wave's heads) ------
  v16bf bw0[4], bw1[4];
  float qv[4];
#pragma unroll
  for (int nl = 0; nl < 4; ++nl) {
    const int ntg = h0 * 2 + nl;
    bw0[nl] = load_b_frag_f32(We, D_EMB, 0,  ntg * 16, lane);
    bw1[nl] = load_b_frag_f32(We, D_EMB, 32, ntg * 16, lane);
    qv[nl]  = qs[ntg * 16 + cl];
  }

  float m_run[2] = {-1e30f, -1e30f};
  float s_run[2] = {0.f, 0.f};
  float o_acc[4] = {0.f, 0.f, 0.f, 0.f};

  for (int phase = 0; phase < 2; ++phase) {
    if (phase == 1) {
      if (wave == 0) __builtin_amdgcn_s_wait_tensorcnt(0);   // second half
      __syncthreads();
    }
    for (int jt = phase * 16; jt < phase * 16 + 16; ++jt) {
      const int j0 = jt * 16;

      // ee tile for this wave's 64 columns: 8 WMMAs, A from LDS
      v16bf a0 = load_a_frag_lds(e_lds, j0, 0,  lane);
      v16bf a1 = load_a_frag_lds(e_lds, j0, 32, lane);
      v8f ee[4];
#pragma unroll
      for (int nl = 0; nl < 4; ++nl) {
        v8f acc = {};
        acc = wmma_bf16(a0, bw0[nl], acc);
        acc = wmma_bf16(a1, bw1[nl], acc);
        ee[nl] = acc;
      }

      // per-column logit partials, K read contiguously from KT
      float part[2][8];
#pragma unroll
      for (int h = 0; h < 2; ++h)
#pragma unroll
        for (int r = 0; r < 8; ++r) part[h][r] = 0.f;

#pragma unroll
      for (int nl = 0; nl < 4; ++nl) {
        const int c = (h0 * 2 + nl) * 16 + cl;
        const float* kc = KT + (size_t)c * SEQ + j0 + rbase;  // 8 contiguous
        const int h = nl >> 1;
#pragma unroll
        for (int r = 0; r < 8; ++r)
          part[h][r] += qv[nl] * (ee[nl][r] + kc[r]);
      }

      // reduce over the 16 lanes (columns) of each half-wave
#pragma unroll
      for (int h = 0; h < 2; ++h)
#pragma unroll
        for (int r = 0; r < 8; ++r) {
          float v = part[h][r];
          v += __shfl_xor(v, 1, 32);
          v += __shfl_xor(v, 2, 32);
          v += __shfl_xor(v, 4, 32);
          v += __shfl_xor(v, 8, 32);
          part[h][r] = v;              // logit(j0+rbase+r, head h0+h)
        }

      // online softmax + value accumulation (V read contiguously from VT)
#pragma unroll
      for (int h = 0; h < 2; ++h) {
        float mt_loc = part[h][0];
#pragma unroll
        for (int r = 1; r < 8; ++r) mt_loc = fmaxf(mt_loc, part[h][r]);
        mt_loc = fmaxf(mt_loc, __shfl_xor(mt_loc, 16, 32));

        const float m_new = fmaxf(m_run[h], mt_loc);
        const float alpha = __expf(m_run[h] - m_new);

        float p[8];
        float psum = 0.f;
#pragma unroll
        for (int r = 0; r < 8; ++r) {
          p[r] = __expf(part[h][r] - m_new);
          psum += p[r];
        }
        psum += __shfl_xor(psum, 16, 32);
        s_run[h] = s_run[h] * alpha + psum;
        m_run[h] = m_new;

#pragma unroll
        for (int nn = 0; nn < 2; ++nn) {
          const int nl = h * 2 + nn;
          const int c  = (h0 * 2 + nl) * 16 + cl;
          const float* vc = VT + (size_t)c * SEQ + j0 + rbase;
          float acc = o_acc[nl] * alpha;
#pragma unroll
          for (int r = 0; r < 8; ++r)
            acc += p[r] * (ee[nl][r] + vc[r]);
          o_acc[nl] = acc;
        }
      }
    }
  }

  // combine the two 8-row lane groups and write out
#pragma unroll
  for (int nl = 0; nl < 4; ++nl) {
    const float o = o_acc[nl] + __shfl_xor(o_acc[nl], 16, 32);
    if (lane < 16) {
      const int c = (h0 * 2 + nl) * 16 + cl;
      Oatt[(size_t)i * D_EMB + c] = o / s_run[nl >> 1];
    }
  }
}

// ---------------------------------------------------------------------------
// Kernel 3: output projection.  Oatt(512x256) @ Wo(256x64) + bo -> 512x64.
// ---------------------------------------------------------------------------
__global__ __launch_bounds__(32)
void out_proj_kernel(const float* __restrict__ A,
                     const float* __restrict__ Wo,
                     const float* __restrict__ bo,
                     float* __restrict__ out) {
  const int lane = threadIdx.x;
  const int mt   = blockIdx.x;
  const int nt   = blockIdx.y;

  v8f acc = {};
#pragma unroll
  for (int ks = 0; ks < D_EMB / 32; ++ks) {
    v16bf a = load_a_frag_f32(A,  D_EMB,  mt * 16, ks * 32, lane);
    v16bf b = load_b_frag_f32(Wo, D_ORIG, ks * 32, nt * 16, lane);
    acc = wmma_bf16(a, b, acc);
  }

  const int col   = nt * 16 + (lane & 15);
  const int rbase = mt * 16 + ((lane >> 4) << 3);
  const float bias = bo[col];
#pragma unroll
  for (int r = 0; r < 8; ++r)
    out[(size_t)(rbase + r) * D_ORIG + col] = acc[r] + bias;
}

// ---------------------------------------------------------------------------
// Launcher.  Inputs: x, e, Wq, Wk, Wv, We, Wo, bo — fp32.  Out: 512x64 fp32.
// Workspace: Q | KT | VT | Oatt (each 512*256 fp32, 2 MB total).
// ---------------------------------------------------------------------------
extern "C" void kernel_launch(void* const* d_in, const int* in_sizes, int n_in,
                              void* d_out, int out_size, void* d_ws,
                              size_t ws_size, hipStream_t stream) {
  (void)in_sizes; (void)n_in; (void)out_size; (void)ws_size;
  const float* x  = (const float*)d_in[0];
  const float* e  = (const float*)d_in[1];
  const float* Wq = (const float*)d_in[2];
  const float* Wk = (const float*)d_in[3];
  const float* Wv = (const float*)d_in[4];
  const float* We = (const float*)d_in[5];
  const float* Wo = (const float*)d_in[6];
  const float* bo = (const float*)d_in[7];
  float* out = (float*)d_out;

  float* Q    = (float*)d_ws;
  float* KT   = Q  + (size_t)SEQ * D_EMB;
  float* VT   = KT + (size_t)SEQ * D_EMB;
  float* Oatt = VT + (size_t)SEQ * D_EMB;

  qkv_proj_kernel<<<dim3(SEQ / 16, D_EMB / 16, 3), 32, 0, stream>>>(
      x, Wq, Wk, Wv, Q, KT, VT);

  const size_t smem_bytes = (SEQ * D_ORIG + D_EMB) * sizeof(float); // 132 KB
  edge_attn_kernel<<<dim3(SEQ), 128, smem_bytes, stream>>>(
      e, We, Q, KT, VT, Oatt);

  out_proj_kernel<<<dim3(SEQ / 16, D_ORIG / 16), 32, 0, stream>>>(
      Oatt, Wo, bo, out);
}
